// SimpleGNN1_16045997817884
// MI455X (gfx1250) — matrix operand order repaired and checked by reading, not compiled, not run on hardware
//
#include <hip/hip_runtime.h>
#include <hip/hip_bf16.h>
#include <hip/amd_detail/amd_hip_unsafe_atomics.h>

#define NN   100000
#define EE_  1600000
#define GG   256
#define FF   128
#define HH   128
#define LL   3

typedef __attribute__((ext_vector_type(16))) __bf16 v16bf;
typedef __attribute__((ext_vector_type(8)))  float  v8f;

// ---------------- helpers ----------------

__device__ __forceinline__ unsigned short f2bf(float f) {
  unsigned int u = __float_as_uint(f);
  unsigned int r = u + 0x7FFFu + ((u >> 16) & 1u);   // round-to-nearest-even
  return (unsigned short)(r >> 16);
}

__device__ __forceinline__ float atomAddF(float* p, float v) {
  return unsafeAtomicAdd(p, v);   // lowers to global_atomic_add_f32
}

__device__ __forceinline__ void atomicMaxF(float* addr, float val) {
  if (val >= 0.0f) atomicMax((int*)addr, __float_as_int(val));
  else             atomicMin((unsigned int*)addr, __float_as_uint(val));
}

// ---------------- GEMM: h = x @ W  (bf16 WMMA, f32 accum) ----------------
// Block = 256 threads = 8 waves. Wave w handles row-tile (blockIdx.x*8+w):
// 16 rows x 128 cols, K=128 -> 8 col-tiles x 4 chained wmma(K=32).
// W staged in LDS as bf16 pre-swizzled into B-fragment lane layout:
// fragment(kt,nt) = 1KB, lane L's 16 halves contiguous at L*32 bytes.
// B 32x16 layout: lane group g = L/16, col N = nt*16 + L%16, elem e<8 -> K = kt*32 + 8*g + e ; e>=8 -> K = kt*32 + 16 + 8*g + (e-8).
__global__ __launch_bounds__(256) void gemm_wmma_bf16(
    const float* __restrict__ X, const float* __restrict__ W,
    float* __restrict__ Hout, int ntiles) {
  __shared__ unsigned short Wlds[16384];   // 32 fragments * 512 halves

  const int t = threadIdx.x;
  // Cooperative fill: 16384 halves / 256 threads = 64 each.
  #pragma unroll 4
  for (int j = 0; j < 64; ++j) {
    int idx  = t * 64 + j;
    int frag = idx >> 9;           // /512
    int kt   = frag >> 3;
    int nt   = frag & 7;
    int rem  = idx & 511;
    int lane = rem >> 4;
    int e    = rem & 15;
    int g    = lane >> 4;
    int K    = kt * 32 + 8 * g + (e & 7) + ((e >> 3) << 4);
    int Ncol = nt * 16 + (lane & 15);
    Wlds[idx] = f2bf(W[K * HH + Ncol]);
  }
  __syncthreads();

  const int wave = t >> 5;
  const int lane = t & 31;
  const int rt   = blockIdx.x * 8 + wave;
  if (rt >= ntiles) return;

  const int g   = lane >> 4;
  const int row = rt * 16 + (lane & 15);
  const float4* xr = (const float4*)(X + (size_t)row * FF);

  // A fragments: 16-bit A 16x32 layout: e<8 -> K=kt*32+8g+e ; e>=8 -> K=kt*32+16+8g+(e-8)
  union AV { v16bf v; unsigned short u[16]; };
  AV a[4];
  #pragma unroll
  for (int kt = 0; kt < 4; ++kt) {
    float4 c0 = xr[kt * 8 + 2 * g];
    float4 c1 = xr[kt * 8 + 2 * g + 1];
    float4 c2 = xr[kt * 8 + 2 * g + 4];
    float4 c3 = xr[kt * 8 + 2 * g + 5];
    a[kt].u[0]=f2bf(c0.x); a[kt].u[1]=f2bf(c0.y); a[kt].u[2]=f2bf(c0.z); a[kt].u[3]=f2bf(c0.w);
    a[kt].u[4]=f2bf(c1.x); a[kt].u[5]=f2bf(c1.y); a[kt].u[6]=f2bf(c1.z); a[kt].u[7]=f2bf(c1.w);
    a[kt].u[8]=f2bf(c2.x); a[kt].u[9]=f2bf(c2.y); a[kt].u[10]=f2bf(c2.z); a[kt].u[11]=f2bf(c2.w);
    a[kt].u[12]=f2bf(c3.x); a[kt].u[13]=f2bf(c3.y); a[kt].u[14]=f2bf(c3.z); a[kt].u[15]=f2bf(c3.w);
  }

  const int mbase = rt * 16 + g * 8;
  const int colb  = lane & 15;
  #pragma unroll
  for (int nt = 0; nt < 8; ++nt) {
    v8f c = {};
    #pragma unroll
    for (int kt = 0; kt < 4; ++kt) {
      const v16bf* bp = (const v16bf*)&Wlds[(kt * 8 + nt) * 512 + lane * 16];
      c = __builtin_amdgcn_wmma_f32_16x16x32_bf16(false, a[kt].v, false, *bp,
                                                  (short)0, c, false, false);
    }
    const int col = nt * 16 + colb;
    #pragma unroll
    for (int r = 0; r < 8; ++r)
      Hout[(size_t)(mbase + r) * HH + col] = c[r];
  }
}

// ---------------- attention scalars: es = h.a_src, ed = h.a_dst ----------------
__global__ __launch_bounds__(256) void attn_k(const float* __restrict__ Hm,
    const float* __restrict__ asrc, const float* __restrict__ adst,
    float* __restrict__ es, float* __restrict__ ed, int n) {
  long tid = (long)blockIdx.x * 256 + threadIdx.x;
  int w = (int)(tid >> 5);
  int lane = threadIdx.x & 31;
  if (w >= n) return;
  float4 h4 = ((const float4*)(Hm + (size_t)w * HH))[lane];
  float4 s4 = ((const float4*)asrc)[lane];
  float4 d4 = ((const float4*)adst)[lane];
  float s = h4.x*s4.x + h4.y*s4.y + h4.z*s4.z + h4.w*s4.w;
  float d = h4.x*d4.x + h4.y*d4.y + h4.z*d4.z + h4.w*d4.w;
  #pragma unroll
  for (int off = 16; off > 0; off >>= 1) {
    s += __shfl_down(s, off, 32);
    d += __shfl_down(d, off, 32);
  }
  if (lane == 0) { es[w] = s; ed[w] = d; }
}

// ---------------- per-layer init ----------------
__global__ __launch_bounds__(256) void layer_init_k(float* __restrict__ agg,
    float* __restrict__ emax, float* __restrict__ denom, long total, int n) {
  long i = (long)blockIdx.x * 256 + threadIdx.x;
  if (i < total) agg[i] = 0.0f;
  if (i < n) { emax[i] = -__builtin_inff(); denom[i] = 0.0f; }
}

// ---------------- edge pass 1: segment max ----------------
__global__ __launch_bounds__(256) void edge_max_k(const int* __restrict__ ei,
    int E, int n, const float* __restrict__ es, const float* __restrict__ ed,
    float* __restrict__ emax) {
  int i = blockIdx.x * 256 + threadIdx.x;
  if (i >= E + n) return;
  int s = (i < E) ? ei[i]     : (i - E);
  int d = (i < E) ? ei[E + i] : (i - E);
  float e = es[s] + ed[d];
  e = (e > 0.0f) ? e : 0.2f * e;            // leaky_relu(0.2)
  atomicMaxF(&emax[d], e);
}

// ---------------- edge pass 2: exp + segment sum ----------------
__global__ __launch_bounds__(256) void edge_exp_k(const int* __restrict__ ei,
    int E, int n, const float* __restrict__ es, const float* __restrict__ ed,
    const float* __restrict__ emax, float* __restrict__ ee,
    float* __restrict__ denom, const float* __restrict__ Hm) {
  int i = blockIdx.x * 256 + threadIdx.x;
  if (i >= E + n) return;
  int s = (i < E) ? ei[i]     : (i - E);
  int d = (i < E) ? ei[E + i] : (i - E);
  float e = es[s] + ed[d];
  e = (e > 0.0f) ? e : 0.2f * e;
  float v = __expf(e - emax[d]);
  ee[i] = v;
  atomAddF(&denom[d], v);
  __builtin_prefetch(Hm + (size_t)s * HH, 0, 1);   // warm L2 for aggregate pass
}

// ---------------- edge pass 3: weighted scatter-aggregate (wave per edge) ----------------
__global__ __launch_bounds__(256) void edge_aggr_k(const int* __restrict__ ei,
    int E, int n, const float* __restrict__ Hm, const float* __restrict__ ee,
    const float* __restrict__ denom, float* __restrict__ agg) {
  long tid = (long)blockIdx.x * 256 + threadIdx.x;
  long w = tid >> 5;
  int lane = threadIdx.x & 31;
  if (w >= (long)E + n) return;
  int i = (int)w;
  int s = (i < E) ? ei[i]     : (i - E);
  int d = (i < E) ? ei[E + i] : (i - E);
  float alpha = ee[i] / denom[d];
  float4 hv = ((const float4*)(Hm + (size_t)s * HH))[lane];
  float* ap = agg + (size_t)d * HH + lane * 4;
  atomAddF(ap + 0, hv.x * alpha);
  atomAddF(ap + 1, hv.y * alpha);
  atomAddF(ap + 2, hv.z * alpha);
  atomAddF(ap + 3, hv.w * alpha);
}

// ---------------- bias + relu (in place -> becomes next x) ----------------
__global__ __launch_bounds__(256) void bias_relu_k(float* __restrict__ agg,
    const float* __restrict__ bias, long total) {
  long i = (long)blockIdx.x * 256 + threadIdx.x;
  if (i >= total) return;
  float v = agg[i] + bias[i & (HH - 1)];
  agg[i] = (v > 0.0f) ? v : 0.0f;
}

// ---------------- pooling ----------------
__global__ __launch_bounds__(256) void pool_init_k(float* __restrict__ sums,
                                                   float* __restrict__ cnt) {
  int i = blockIdx.x * 256 + threadIdx.x;
  if (i < GG * HH) sums[i] = 0.0f;
  if (i < GG) cnt[i] = 0.0f;
}

__global__ __launch_bounds__(256) void pool_accum_k(const float* __restrict__ x,
    const int* __restrict__ bids, float* __restrict__ sums,
    float* __restrict__ cnt, long total) {
  long i = (long)blockIdx.x * 256 + threadIdx.x;
  if (i >= total) return;
  int node = (int)(i >> 7);
  int c = (int)(i & 127);
  int g = bids[node];
  atomAddF(&sums[(size_t)g * HH + c], x[i]);
  if (c == 0) atomAddF(&cnt[g], 1.0f);
}

// ---------------- heads: BN + FC1/FC2/FC3 ----------------
__global__ __launch_bounds__(128) void head_k(const float* __restrict__ sums,
    const float* __restrict__ cnt, const float* __restrict__ gfeat,
    const float* __restrict__ bne, const float* __restrict__ bnt1,
    const float* __restrict__ bnc,
    const float* __restrict__ f1w, const float* __restrict__ f1b,
    const float* __restrict__ f2w, const float* __restrict__ f2b,
    const float* __restrict__ f3aw, const float* __restrict__ f3ab,
    const float* __restrict__ f3bw, const float* __restrict__ f3bb,
    float* __restrict__ out) {
  __shared__ float xs[128];
  __shared__ float x3[133];
  __shared__ float o2[2];
  const int g = blockIdx.x, t = threadIdx.x;

  float c = fmaxf(cnt[g], 1.0f);
  float xm = sums[(size_t)g * HH + t] / c;
  float xb = (xm - bne[256 + t]) * bne[t] * rsqrtf(bne[384 + t] + 1e-5f) + bne[128 + t];
  xs[t] = xb;

  const float* gf = gfeat + g * 8;
  const float lev = gf[7], nx0 = gf[5], nx1 = gf[6], nx2 = gf[7];
  const float wf1 = gf[2], wf2 = gf[1];

  // bn_comb over [xs(128), wf1, wf2, nx0, nx1, nx2]
  x3[t] = (xb - bnc[266 + t]) * bnc[t] * rsqrtf(bnc[399 + t] + 1e-5f) + bnc[133 + t];
  if (t < 5) {
    float ext = (t == 0) ? wf1 : (t == 1) ? wf2 : (t == 2) ? nx0 : (t == 3) ? nx1 : nx2;
    int idx = 128 + t;
    x3[idx] = (ext - bnc[266 + idx]) * bnc[idx] * rsqrtf(bnc[399 + idx] + 1e-5f) + bnc[133 + idx];
  }
  __syncthreads();

  if (t < 10) {                    // out1 = bn_t1(relu([xs,lev] @ fc1))
    float acc = f1b[t];
    for (int k = 0; k < 128; ++k) acc += xs[k] * f1w[k * 10 + t];
    acc += lev * f1w[128 * 10 + t];
    acc = fmaxf(acc, 0.0f);
    acc = (acc - bnt1[20 + t]) * bnt1[t] * rsqrtf(bnt1[30 + t] + 1e-5f) + bnt1[10 + t];
    out[g * 10 + t] = acc;
  }
  if (t < 2) {                     // out2 = relu([xs,nx] @ fc2)
    float acc = f2b[t];
    for (int k = 0; k < 128; ++k) acc += xs[k] * f2w[k * 2 + t];
    acc += nx0 * f2w[128 * 2 + t] + nx1 * f2w[129 * 2 + t] + nx2 * f2w[130 * 2 + t];
    acc = fmaxf(acc, 0.0f);
    o2[t] = acc;
    out[GG * 10 + g * 2 + t] = acc;
  }
  __syncthreads();

  int pred = (o2[1] > o2[0]) ? 1 : 0;   // jnp.argmax (first max)
  float m0 = (pred == 0) ? 1.0f : 0.0f;
  float m1 = (pred == 1) ? 1.0f : 0.0f;
  const int o3base = GG * 10 + GG * 2;
  if (t < 4) {
    float acc = f3ab[t];
    for (int k = 0; k < 133; ++k) acc += x3[k] * f3aw[k * 4 + t];
    out[o3base + g * 9 + t] = acc * m0;
  } else if (t < 9) {
    int j = t - 4;
    float acc = f3bb[j];
    for (int k = 0; k < 133; ++k) acc += x3[k] * f3bw[k * 5 + j];
    out[o3base + g * 9 + 4 + j] = acc * m1;
  }
}

// ---------------- host launcher ----------------
extern "C" void kernel_launch(void* const* d_in, const int* in_sizes, int n_in,
                              void* d_out, int out_size, void* d_ws, size_t ws_size,
                              hipStream_t stream) {
  const float* x_in  = (const float*)d_in[0];
  const int*   eidx  = (const int*)d_in[1];
  const int*   bids  = (const int*)d_in[2];
  const float* gfeat = (const float*)d_in[3];
  const float* gatW  = (const float*)d_in[5];
  const float* asrc  = (const float*)d_in[6];
  const float* adst  = (const float*)d_in[7];
  const float* gbias = (const float*)d_in[8];
  const float* bne   = (const float*)d_in[9];
  const float* bnt1  = (const float*)d_in[10];
  const float* bnc   = (const float*)d_in[11];
  const float* f1w   = (const float*)d_in[12];
  const float* f1b   = (const float*)d_in[13];
  const float* f2w   = (const float*)d_in[14];
  const float* f2b   = (const float*)d_in[15];
  const float* f3aw  = (const float*)d_in[16];
  const float* f3ab  = (const float*)d_in[17];
  const float* f3bw  = (const float*)d_in[18];
  const float* f3bb  = (const float*)d_in[19];
  float* out = (float*)d_out;

  const size_t NF = (size_t)NN * HH;
  float* ws    = (float*)d_ws;
  float* bufA  = ws;
  float* bufB  = ws + NF;
  float* bufH  = ws + 2 * NF;
  float* es    = ws + 3 * NF;
  float* ed    = es + NN;
  float* emax  = ed + NN;
  float* denom = emax + NN;
  float* ee    = denom + NN;           // EE_+NN
  float* sums  = ee + (EE_ + NN);
  float* cnt   = sums + (size_t)GG * HH;

  const int ntiles   = NN / 16;                       // 6250
  const int gemmBlk  = (ntiles + 7) / 8;              // 782
  const int attnBlk  = (NN * 32 + 255) / 256;         // 12500
  const long totNF   = (long)NN * HH;
  const int initBlk  = (int)((totNF + 255) / 256);    // 50000
  const int edges    = EE_ + NN;
  const int edgeBlk  = (edges + 255) / 256;           // 6641
  const int aggrBlk  = (int)(((long)edges * 32 + 255) / 256);  // 212500

  const float* xcur = x_in;
  for (int l = 0; l < LL; ++l) {
    float* agg = (l == 0) ? bufA : (l == 1) ? bufB : bufA;
    const float* Wl = gatW + (size_t)l * FF * HH;

    gemm_wmma_bf16<<<gemmBlk, 256, 0, stream>>>(xcur, Wl, bufH, ntiles);
    attn_k<<<attnBlk, 256, 0, stream>>>(bufH, asrc + l * HH, adst + l * HH, es, ed, NN);
    layer_init_k<<<initBlk, 256, 0, stream>>>(agg, emax, denom, totNF, NN);
    edge_max_k<<<edgeBlk, 256, 0, stream>>>(eidx, EE_, NN, es, ed, emax);
    edge_exp_k<<<edgeBlk, 256, 0, stream>>>(eidx, EE_, NN, es, ed, emax, ee, denom, bufH);
    edge_aggr_k<<<aggrBlk, 256, 0, stream>>>(eidx, EE_, NN, bufH, ee, denom, agg);
    bias_relu_k<<<initBlk, 256, 0, stream>>>(agg, gbias + l * HH, totNF);
    xcur = agg;
  }

  pool_init_k<<<(GG * HH + 255) / 256, 256, 0, stream>>>(sums, cnt);
  pool_accum_k<<<initBlk, 256, 0, stream>>>(xcur, bids, sums, cnt, totNF);
  head_k<<<GG, 128, 0, stream>>>(sums, cnt, gfeat, bne, bnt1, bnc,
                                 f1w, f1b, f2w, f2b, f3aw, f3ab, f3bw, f3bb, out);
}
